// RecurrentGlimpseDecoder_11811160064307
// MI455X (gfx1250) — matrix operand order repaired and checked by reading, not compiled, not run on hardware
//
#include <hip/hip_runtime.h>
#include <hip/hip_bf16.h>
#include <cstdint>

// ---------------------------------------------------------------------------
// RecurrentGlimpseDecoder for MI455X (gfx1250, wave32, WMMA)
//
// Phase 1: f32->f16 conversion + graph mean.
// Phase 2: WMMA GEMM  E(65536x128) @ W^T(128x128) for K_logit / Kh / Vh,
//          using v_wmma_f32_16x16x32_f16 (f32 accumulate), f16 outputs.
// Phase 3: persistent decode kernel: 1 workgroup (256 thr = 8 waves) per
//          batch element runs all 255 greedy steps; Kh/Vh live in LDS (f16,
//          padded strides for bank-conflict-free access), weights + K_logit
//          stream from L2 (192 MB -> fully resident). The static 2/3 of the
//          context projection is hoisted out of the step loop. Kh staging
//          uses GLOBAL_LOAD_ASYNC_TO_LDS_B64 (ASYNCcnt-tracked).
// ---------------------------------------------------------------------------

#define BS_ 256
#define N_  256
#define D_  128
#define H_  8
#define T_  4
#define R_  8

typedef _Float16 half8  __attribute__((ext_vector_type(8)));
typedef _Float16 half16 __attribute__((ext_vector_type(16)));
typedef float    float8 __attribute__((ext_vector_type(8)));
typedef int      v2i_t  __attribute__((ext_vector_type(2)));

#if __has_builtin(__builtin_amdgcn_global_load_async_to_lds_b64) && \
    __has_builtin(__builtin_amdgcn_s_wait_asynccnt)
#define HAVE_ASYNC_LDS 1
#else
#define HAVE_ASYNC_LDS 0
#endif

// ------------------------------ converters ---------------------------------

__global__ __launch_bounds__(256) void cvt_f32_f16(const float* __restrict__ s,
                                                   _Float16* __restrict__ d, int cnt) {
  int i = blockIdx.x * 256 + threadIdx.x;
  if (i < cnt) d[i] = (_Float16)s[i];
}

__global__ __launch_bounds__(128) void graph_mean_kernel(const float* __restrict__ E,
                                                         float* __restrict__ G) {
  int b = blockIdx.x, d = threadIdx.x;
  float s = 0.f;
  for (int n = 0; n < N_; ++n) s += E[((size_t)b * N_ + n) * D_ + d];
  G[(size_t)b * D_ + d] = s * (1.0f / N_);
}

// ------------------------------ WMMA GEMM ----------------------------------
// Computes O = E @ W^T for three weights. One wave computes a 16-row strip
// (16 x 128 output) with 8 tiles of 16x16, K=128 in 4 chunks of 32.
// A layout (16-bit A 16x32): lanes 0-15 row=lane, K=0..15 contiguous;
//                            lanes 16-31 row=lane-16, K=16..31.
// B layout (16-bit B 32x16): lanes 0-15 col=lane, K=k0..k0+15 contiguous;
//                            lanes 16-31 col=lane-16, K=k0+16..k0+31.
// Since (E@W^T)[m,n] = sum_k E[m,k]*W[n,k], B column n is W row n (row-major).

__global__ __launch_bounds__(256) void wmma_proj_kernel(
    const _Float16* __restrict__ E,
    const _Float16* __restrict__ W0, const _Float16* __restrict__ W1,
    const _Float16* __restrict__ W2,
    _Float16* __restrict__ O0, _Float16* __restrict__ O1,
    _Float16* __restrict__ O2) {
  const int which = blockIdx.y;
  const _Float16* W = (which == 0) ? W0 : (which == 1) ? W1 : W2;
  _Float16* O = (which == 0) ? O0 : (which == 1) ? O1 : O2;

  const int lane = threadIdx.x & 31;
  const int wave = threadIdx.x >> 5;
  const int m0 = blockIdx.x * 128 + wave * 16;

  // --- preload A fragments for all 4 K-chunks ---
  const int arow = m0 + (lane & 15);
  const int akb = (lane < 16) ? 0 : 16;
  const _Float16* abase = E + (size_t)arow * D_;
  half16 afrag[4];
#pragma unroll
  for (int kc = 0; kc < 4; ++kc) {
    half8 lo = *(const half8*)(abase + kc * 32 + akb);
    half8 hi = *(const half8*)(abase + kc * 32 + akb + 8);
#pragma unroll
    for (int i = 0; i < 8; ++i) { afrag[kc][i] = lo[i]; afrag[kc][i + 8] = hi[i]; }
  }

  const int ncol = lane & 15;
  const int bko = (lane < 16) ? 0 : 16;
  const int rowbase = m0 + ((lane < 16) ? 0 : 8);

#pragma unroll
  for (int nt = 0; nt < 8; ++nt) {
    const _Float16* bbase = W + (size_t)(nt * 16 + ncol) * D_ + bko;
    // stage all four B fragments first so the loads pipeline (incremental
    // s_wait_loadcnt instead of a full wait before every WMMA)
    half16 bfrag[4];
#pragma unroll
    for (int kc = 0; kc < 4; ++kc) bfrag[kc] = *(const half16*)(bbase + kc * 32);
    float8 c = {0.f, 0.f, 0.f, 0.f, 0.f, 0.f, 0.f, 0.f};
#pragma unroll
    for (int kc = 0; kc < 4; ++kc) {
      c = __builtin_amdgcn_wmma_f32_16x16x32_f16(false, afrag[kc], false, bfrag[kc],
                                                 (short)0, c, false, false);
    }
    _Float16* orow = O + (size_t)rowbase * D_ + nt * 16 + ncol;
#pragma unroll
    for (int i = 0; i < 8; ++i) orow[(size_t)i * D_] = (_Float16)c[i];
  }
}

// ------------------------------ decode kernel ------------------------------

__global__ __launch_bounds__(256) void decode_kernel(
    const float* __restrict__ emb,        // (BS,N,D) f32
    const float* __restrict__ graph,      // (BS,D)   f32
    const _Float16* __restrict__ KhF,     // (BS,N,D) f16
    const _Float16* __restrict__ VhF,     // (BS,N,D) f16
    const _Float16* __restrict__ KlF,     // (BS,N,D) f16
    const _Float16* __restrict__ WctxF,   // (D,3D)   f16
    const _Float16* __restrict__ WoutF,   // (D,D)    f16
    const float* __restrict__ loraA,      // (T,R,D)  f32
    const float* __restrict__ loraB,      // (T,D,R)  f32
    int* __restrict__ tours,              // (BS,N)
    float* __restrict__ lp_out) {         // (BS)
  constexpr int KP = 132;  // padded row stride (halves) for sK
  constexpr int VP = 258;  // padded row stride (halves) for transposed sV

  extern __shared__ char smem[];
  _Float16* sK = (_Float16*)smem;                 // 256 x KP
  _Float16* sV = sK + 256 * KP;                   // 128 x VP (transposed V)
  float* fb = (float*)(smem + (size_t)(256 * KP + 128 * VP) * 2);
  float* sCtx = fb;  fb += 384;
  float* sQs = fb;   fb += 128;   // hoisted static part of q
  float* sQ = fb;    fb += 128;
  float* sAttn = fb; fb += 8 * 256;
  float* sAO = fb;   fb += 128;
  float* sPart = fb; fb += 256;
  float* sRedV = fb; fb += 8;
  float* sSum = fb;  fb += 8;
  float* sLa = fb;   fb += 8;
  float* sMisc = fb; fb += 4;  // [0]=row max, [1]=log-prob accum
  int* ib = (int*)fb;
  int* sRedI = ib;          ib += 8;
  unsigned* sVis = (unsigned*)ib; ib += 8;
  int* sPrev = ib;          ib += 1;
  int* sNext = ib;          ib += 1;

  const int tid = threadIdx.x;
  const int b = blockIdx.x;
  const int lane = tid & 31;
  const int wv = tid >> 5;
  const float NEG_INF = -__builtin_inff();

  // ---- stage Kh (padded) and Vh (transposed) into LDS ----
  {
    const uint2* gK2 = (const uint2*)(KhF + (size_t)b * N_ * D_);
#if HAVE_ASYNC_LDS
    // async global->LDS DMA (ASYNCcnt-tracked), <=32 outstanding per wave
    for (int idx = tid; idx < N_ * 32; idx += 256) {
      int n = idx >> 5, w = idx & 31;
      __builtin_amdgcn_global_load_async_to_lds_b64(
          (__attribute__((address_space(1))) v2i_t*)(gK2 + n * 32 + w),
          (__attribute__((address_space(3))) v2i_t*)((char*)sK +
                                                     (size_t)n * (KP * 2) + w * 8),
          0, 0);
    }
#else
    for (int idx = tid; idx < N_ * 32; idx += 256) {
      int n = idx >> 5, w = idx & 31;
      *(uint2*)((char*)sK + (size_t)n * (KP * 2) + w * 8) = gK2[n * 32 + w];
    }
#endif
    const _Float16* gV = VhF + (size_t)b * N_ * D_;
    for (int idx = tid; idx < N_ * D_; idx += 256) {
      int n = idx >> 7, d = idx & 127;
      sV[d * VP + n] = gV[n * D_ + d];
    }
  }
  if (tid < 128) {
    sCtx[tid] = graph[(size_t)b * D_ + tid];
    sCtx[128 + tid] = emb[((size_t)b * N_ + 0) * D_ + tid];  // first city = 0
  }
  if (tid < 8) sVis[tid] = (tid == 0) ? 1u : 0u;
  if (tid == 0) {
    sPrev[0] = 0;
    sMisc[1] = 0.f;
    tours[(size_t)b * N_ + 0] = 0;
  }
  // thread tid always scores city `tid`: warm its K_logit row (global_prefetch_b8)
  __builtin_prefetch(KlF + ((size_t)b * N_ + tid) * D_, 0, 1);
#if HAVE_ASYNC_LDS
  __builtin_amdgcn_s_wait_asynccnt(0);
#endif
  __syncthreads();

  // ---- hoisted static context projection: q_static = [graph,first] @ Wctx[:, :256]^T
  if (tid < 128) {
    const half8* wr = (const half8*)(WctxF + (size_t)tid * 384);
    float acc = 0.f;
#pragma unroll 4
    for (int c = 0; c < 32; ++c) {
      half8 w8 = wr[c];
#pragma unroll
      for (int i = 0; i < 8; ++i) acc += sCtx[c * 8 + i] * (float)w8[i];
    }
    sQs[tid] = acc;
  }
  __syncthreads();

  const float inv_sqrt_dk = 0.25f;                 // 1/sqrt(16)
  const float inv_sqrt_d = 0.08838834764831845f;   // 1/sqrt(128)

  for (int step = 0; step < N_ - 1; ++step) {
    // ctx[256:384] = embeddings[b, prev]
    {
      int prev = sPrev[0];
      if (tid < 128) sCtx[256 + tid] = emb[((size_t)b * N_ + prev) * D_ + tid];
    }
    __syncthreads();

    // q = q_static + prev_embed @ Wctx[:, 256:384]^T  (per-step: 128x128 only)
    if (tid < 128) {
      const half8* wr = (const half8*)(WctxF + (size_t)tid * 384 + 256);
      float acc = sQs[tid];
#pragma unroll 4
      for (int c = 0; c < 16; ++c) {
        half8 w8 = wr[c];
#pragma unroll
        for (int i = 0; i < 8; ++i) acc += sCtx[256 + c * 8 + i] * (float)w8[i];
      }
      sQ[tid] = acc;
    }
    __syncthreads();

    // ---- T thinking iterations ----
    for (int it = 0; it < T_; ++it) {
      // scores + per-head softmax: wave wv owns head wv (n = lane + 32*j)
      {
        float qh[16];
#pragma unroll
        for (int i = 0; i < 16; ++i) qh[i] = sQ[wv * 16 + i];
        float sl[8];
        float mx = NEG_INF;
#pragma unroll
        for (int j = 0; j < 8; ++j) {
          int n = lane + 32 * j;
          const unsigned* kr =
              (const unsigned*)((char*)sK + (size_t)n * (KP * 2) + wv * 32);
          float s = 0.f;
#pragma unroll
          for (int i = 0; i < 8; ++i) {
            union { unsigned u; _Float16 h[2]; } cv;
            cv.u = kr[i];
            s += qh[2 * i] * (float)cv.h[0] + qh[2 * i + 1] * (float)cv.h[1];
          }
          s *= inv_sqrt_dk;
          if ((sVis[n >> 5] >> (n & 31)) & 1u) s = NEG_INF;
          sl[j] = s;
          mx = fmaxf(mx, s);
        }
#pragma unroll
        for (int off = 16; off > 0; off >>= 1) mx = fmaxf(mx, __shfl_xor(mx, off, 32));
        float sum = 0.f;
#pragma unroll
        for (int j = 0; j < 8; ++j) { float e = __expf(sl[j] - mx); sl[j] = e; sum += e; }
#pragma unroll
        for (int off = 16; off > 0; off >>= 1) sum += __shfl_xor(sum, off, 32);
        float inv = 1.f / sum;
#pragma unroll
        for (int j = 0; j < 8; ++j) sAttn[wv * 256 + lane + 32 * j] = sl[j] * inv;
      }
      __syncthreads();

      // attn_out[d] = sum_n attn[h(d),n] * V[n,d]   (2 half-range partials/d)
      {
        int d = tid & 127, hh = tid >> 7;
        int head = d >> 4;
        const unsigned* vr =
            (const unsigned*)((char*)sV + (size_t)d * (VP * 2) + hh * 256);
        const float* ar = sAttn + head * 256 + hh * 128;
        float acc = 0.f;
#pragma unroll 8
        for (int n2 = 0; n2 < 64; ++n2) {
          union { unsigned u; _Float16 h[2]; } cv;
          cv.u = vr[n2];
          acc += ar[2 * n2] * (float)cv.h[0] + ar[2 * n2 + 1] * (float)cv.h[1];
        }
        sPart[tid] = acc;
      }
      __syncthreads();
      if (tid < 128) sAO[tid] = sPart[tid] + sPart[tid + 128];
      __syncthreads();

      // LoRA down-projection (R=8)
      if (tid < 8) {
        const float* arow = loraA + ((size_t)it * R_ + tid) * D_;
        float acc = 0.f;
        for (int k = 0; k < D_; ++k) acc += sAO[k] * arow[k];
        sLa[tid] = acc;
      }
      __syncthreads();

      // q += attn_out @ Wout^T + la @ loraB^T
      if (tid < 128) {
        const half8* wr = (const half8*)(WoutF + (size_t)tid * D_);
        float acc = 0.f;
#pragma unroll 4
        for (int c = 0; c < 16; ++c) {
          half8 w8 = wr[c];
#pragma unroll
          for (int i = 0; i < 8; ++i) acc += sAO[c * 8 + i] * (float)w8[i];
        }
        const float* brow = loraB + ((size_t)it * D_ + tid) * R_;
        float dl = 0.f;
#pragma unroll
        for (int r = 0; r < R_; ++r) dl += sLa[r] * brow[r];
        sQ[tid] += acc + dl;
      }
      __syncthreads();
    }

    // ---- logits, clipped tanh, masked softmax, greedy argmax ----
    {
      const half8* kv = (const half8*)(KlF + ((size_t)b * N_ + tid) * D_);
      float lg = 0.f;
#pragma unroll 4
      for (int c = 0; c < 16; ++c) {
        half8 w8 = kv[c];
#pragma unroll
        for (int i = 0; i < 8; ++i) lg += sQ[c * 8 + i] * (float)w8[i];
      }
      lg = 10.f * tanhf(lg * inv_sqrt_d);
      if ((sVis[tid >> 5] >> (tid & 31)) & 1u) lg = NEG_INF;

      // wave (val,idx) argmax with first-index tie-break
      float v = lg;
      int bi = tid;
#pragma unroll
      for (int off = 16; off > 0; off >>= 1) {
        float ov = __shfl_xor(v, off, 32);
        int oi = __shfl_xor(bi, off, 32);
        if (ov > v || (ov == v && oi < bi)) { v = ov; bi = oi; }
      }
      if (lane == 0) { sRedV[wv] = v; sRedI[wv] = bi; }
      __syncthreads();
      if (tid == 0) {
        float bv = sRedV[0];
        int bidx = sRedI[0];
        for (int w2 = 1; w2 < 8; ++w2) {
          if (sRedV[w2] > bv || (sRedV[w2] == bv && sRedI[w2] < bidx)) {
            bv = sRedV[w2]; bidx = sRedI[w2];
          }
        }
        sMisc[0] = bv;
        sNext[0] = bidx;
      }
      __syncthreads();

      float e = __expf(lg - sMisc[0]);  // masked rows -> exp(-inf) = 0
#pragma unroll
      for (int off = 16; off > 0; off >>= 1) e += __shfl_xor(e, off, 32);
      if (lane == 0) sSum[wv] = e;
      __syncthreads();
      if (tid == 0) {
        float S = 0.f;
        for (int w2 = 0; w2 < 8; ++w2) S += sSum[w2];
        int nx = sNext[0];
        // chosen prob = exp(l_max - max)/S = 1/S
        sMisc[1] += logf(1.f / S + 1e-10f);
        sVis[nx >> 5] |= (1u << (nx & 31));
        sPrev[0] = nx;
        tours[(size_t)b * N_ + step + 1] = nx;
      }
      __syncthreads();
    }
  }
  if (tid == 0) lp_out[b] = sMisc[1];
}

// ------------------------------ launch -------------------------------------

extern "C" void kernel_launch(void* const* d_in, const int* in_sizes, int n_in,
                              void* d_out, int out_size, void* d_ws, size_t ws_size,
                              hipStream_t stream) {
  (void)in_sizes; (void)n_in; (void)out_size; (void)ws_size;

  const float* emb   = (const float*)d_in[0];  // (256,256,128)
  const float* Wctx  = (const float*)d_in[1];  // (128,384)
  const float* Wgk   = (const float*)d_in[2];  // (128,128)
  const float* Wgv   = (const float*)d_in[3];  // (128,128)
  const float* Wgo   = (const float*)d_in[4];  // (128,128)
  const float* Wlk   = (const float*)d_in[5];  // (128,128)
  const float* loraA = (const float*)d_in[6];  // (4,8,128)
  const float* loraB = (const float*)d_in[7];  // (4,128,8)

  // workspace layout (256B aligned blocks)
  char* ws = (char*)d_ws;
  size_t off = 0;
  auto take = [&](size_t bytes) {
    char* p = ws + off;
    off = (off + bytes + 255) & ~(size_t)255;
    return p;
  };
  const size_t EMB_ELEMS = (size_t)BS_ * N_ * D_;  // 8,388,608
  _Float16* Ef     = (_Float16*)take(EMB_ELEMS * 2);
  _Float16* KlF    = (_Float16*)take(EMB_ELEMS * 2);
  _Float16* KhF    = (_Float16*)take(EMB_ELEMS * 2);
  _Float16* VhF    = (_Float16*)take(EMB_ELEMS * 2);
  _Float16* WctxF  = (_Float16*)take((size_t)D_ * 3 * D_ * 2);
  _Float16* WgkF   = (_Float16*)take((size_t)D_ * D_ * 2);
  _Float16* WgvF   = (_Float16*)take((size_t)D_ * D_ * 2);
  _Float16* WgoF   = (_Float16*)take((size_t)D_ * D_ * 2);
  _Float16* WlkF   = (_Float16*)take((size_t)D_ * D_ * 2);
  float* graphE    = (float*)take((size_t)BS_ * D_ * 4);

  // Phase 1: conversions + graph mean
  cvt_f32_f16<<<(int)(EMB_ELEMS / 256), 256, 0, stream>>>(emb, Ef, (int)EMB_ELEMS);
  cvt_f32_f16<<<192, 256, 0, stream>>>(Wctx, WctxF, D_ * 3 * D_);
  cvt_f32_f16<<<64, 256, 0, stream>>>(Wgk, WgkF, D_ * D_);
  cvt_f32_f16<<<64, 256, 0, stream>>>(Wgv, WgvF, D_ * D_);
  cvt_f32_f16<<<64, 256, 0, stream>>>(Wgo, WgoF, D_ * D_);
  cvt_f32_f16<<<64, 256, 0, stream>>>(Wlk, WlkF, D_ * D_);
  graph_mean_kernel<<<BS_, 128, 0, stream>>>(emb, graphE);

  // Phase 2: WMMA projections (y: 0->K_logit, 1->Kh, 2->Vh)
  wmma_proj_kernel<<<dim3(512, 3), 256, 0, stream>>>(Ef, WlkF, WgkF, WgvF,
                                                     KlF, KhF, VhF);

  // Phase 3: persistent per-batch greedy decode
  constexpr int KP = 132, VP = 258;
  size_t smem = (size_t)(256 * KP + 128 * VP) * 2 +
                (size_t)(384 + 128 + 128 + 8 * 256 + 128 + 256 + 8 + 8 + 8 + 4) * 4 +
                (size_t)(8 + 8 + 2) * 4;
  int* tours = (int*)d_out;                      // (256,256) int32
  float* lp = (float*)d_out + (size_t)BS_ * N_;  // (256) f32
  decode_kernel<<<BS_, 256, smem, stream>>>(emb, graphE, KhF, VhF, KlF, WctxF,
                                            WgoF, loraA, loraB, tours, lp);
}
